// _Model_40982577938987
// MI455X (gfx1250) — matrix operand order repaired and checked by reading, not compiled, not run on hardware
//
#include <hip/hip_runtime.h>
#include <hip/hip_bf16.h>

typedef _Float16 half_t;
typedef _Float16 v16h __attribute__((ext_vector_type(16)));
typedef _Float16 v8h  __attribute__((ext_vector_type(8)));
typedef float    v8f  __attribute__((ext_vector_type(8)));

// ---------------------------------------------------------------------------
// WMMA fragment helpers (CDNA5 v_wmma_f32_16x16x32_f16, wave32)
// A 16x32 f16 layout: lane<16 -> row=lane, K {0..7,16..23}; lane>=16 -> K {8..15,24..31}
// B 32x16 f16 layout: col=lane&15; lane<16 -> K {0..7,16..23}; lane>=16 -> K {8..15,24..31}
// C/D 16x16 f32: VGPR r, lane L -> row = r + 8*(L>>4), col = L&15
// ---------------------------------------------------------------------------
static __device__ __forceinline__ v16h load_a_frag(const half_t* base, int stride) {
  const int lane = threadIdx.x & 31;
  const int row  = lane & 15;
  const int hi   = lane >> 4;
  const half_t* p = base + row * stride;
  v8h a0 = *(const v8h*)(p + (hi ? 8  : 0));
  v8h a1 = *(const v8h*)(p + (hi ? 24 : 16));
  return __builtin_shufflevector(a0, a1, 0,1,2,3,4,5,6,7,8,9,10,11,12,13,14,15);
}

static __device__ __forceinline__ v16h load_b_frag(const half_t* base, int stride) {
  const int lane = threadIdx.x & 31;
  const int col  = lane & 15;
  const int hi   = lane >> 4;
  const half_t* p = base + (size_t)col * stride + (hi ? 8 : 0);
  v8h b0 = *(const v8h*)(p);
  v8h b1 = *(const v8h*)(p + 16);
  return __builtin_shufflevector(b0, b1, 0,1,2,3,4,5,6,7,8,9,10,11,12,13,14,15);
}

static __device__ __forceinline__ v8f wmma_step(v16h a, v16h b, v8f c) {
  return __builtin_amdgcn_wmma_f32_16x16x32_f16(false, a, false, b, (short)0, c, false, false);
}

// ---------------------------------------------------------------------------
// Weight convert: f32 [O,I] -> f16 [Op,Ip] zero-padded (row-major)
// ---------------------------------------------------------------------------
__global__ void convert_w_kernel(const float* __restrict__ w, half_t* __restrict__ out,
                                 int O, int I, int Op, int Ip) {
  int t = blockIdx.x * blockDim.x + threadIdx.x;
  if (t >= Op * Ip) return;
  int o = t / Ip, i = t % Ip;
  out[t] = (o < O && i < I) ? (half_t)w[(size_t)o * I + i] : (half_t)0.f;
}

// ---------------------------------------------------------------------------
// Farthest point sampling: one block per batch, NPTS = PPT*256 points in LDS
// Matches jax.lax.scan semantics: emits `last` (pre-update), first index = 0.
// ---------------------------------------------------------------------------
template<int PPT>
__global__ __launch_bounds__(256) void fps_kernel(const float* __restrict__ xyz,
                                                  int npoint, int* __restrict__ idx_out) {
  constexpr int T = 256;
  constexpr int NPTS = PPT * T;
  __shared__ float px[NPTS], py[NPTS], pz[NPTS];
  __shared__ float redv[T];
  __shared__ int   redi[T];
  __shared__ int   s_cur;
  const int b = blockIdx.x, tid = threadIdx.x;
  for (int i = tid; i < NPTS; i += T) {
    size_t o = ((size_t)b * NPTS + i) * 3;
    px[i] = xyz[o + 0]; py[i] = xyz[o + 1]; pz[i] = xyz[o + 2];
  }
  float mind[PPT];
#pragma unroll
  for (int j = 0; j < PPT; ++j) mind[j] = 1e10f;
  if (tid == 0) s_cur = 0;
  __syncthreads();
  for (int it = 0; it < npoint; ++it) {
    int cur = s_cur;
    if (tid == 0) idx_out[b * npoint + it] = cur;
    float cx = px[cur], cy = py[cur], cz = pz[cur];
    float bv = -1.f; int bi = 0;
#pragma unroll
    for (int j = 0; j < PPT; ++j) {
      int i = tid * PPT + j;
      float dx = px[i] - cx, dy = py[i] - cy, dz = pz[i] - cz;
      float d = dx * dx + dy * dy + dz * dz;
      float m = fminf(mind[j], d);
      mind[j] = m;
      if (m > bv) { bv = m; bi = i; }
    }
    redv[tid] = bv; redi[tid] = bi;
    __syncthreads();
    for (int sft = T / 2; sft > 0; sft >>= 1) {
      if (tid < sft) {
        float ov = redv[tid + sft]; int oi = redi[tid + sft];
        if (ov > redv[tid] || (ov == redv[tid] && oi < redi[tid])) {
          redv[tid] = ov; redi[tid] = oi;
        }
      }
      __syncthreads();
    }
    if (tid == 0) s_cur = redi[0];
    __syncthreads();
  }
}

__global__ void gather_xyz_kernel(const float* __restrict__ xyz, const int* __restrict__ idx,
                                  float* __restrict__ out, int n, int S) {
  int t = blockIdx.x * blockDim.x + threadIdx.x;
  if (t >= 32 * S) return;
  int b = t / S;
  int p = idx[t];
  size_t src = ((size_t)b * n + p) * 3;
  out[(size_t)t * 3 + 0] = xyz[src + 0];
  out[(size_t)t * 3 + 1] = xyz[src + 1];
  out[(size_t)t * 3 + 2] = xyz[src + 2];
}

// ---------------------------------------------------------------------------
// Ball query: block per batch, points in LDS, one thread per center scanning
// in index order (== reference sort-by-index + truncate; pad with first hit).
// ---------------------------------------------------------------------------
template<int NPTS>
__global__ __launch_bounds__(256) void ball_query_kernel(const float* __restrict__ new_xyz,
                                                         const float* __restrict__ xyz,
                                                         int S, int nsample, float r2,
                                                         int* __restrict__ idx_out) {
  __shared__ float px[NPTS], py[NPTS], pz[NPTS];
  const int b = blockIdx.x;
  for (int i = threadIdx.x; i < NPTS; i += blockDim.x) {
    size_t o = ((size_t)b * NPTS + i) * 3;
    px[i] = xyz[o + 0]; py[i] = xyz[o + 1]; pz[i] = xyz[o + 2];
  }
  __syncthreads();
  for (int s = threadIdx.x; s < S; s += blockDim.x) {
    size_t o = (size_t)b * S + s;
    float cx = new_xyz[o * 3 + 0], cy = new_xyz[o * 3 + 1], cz = new_xyz[o * 3 + 2];
    int* op = idx_out + o * nsample;
    int cnt = 0, first = 0;
    bool got = false;
    for (int i = 0; i < NPTS && cnt < nsample; ++i) {
      float dx = px[i] - cx, dy = py[i] - cy, dz = pz[i] - cz;
      if (dx * dx + dy * dy + dz * dz <= r2) {
        if (!got) { first = i; got = true; }
        op[cnt++] = i;
      }
    }
    for (; cnt < nsample; ++cnt) op[cnt] = first;
  }
}

// ---------------------------------------------------------------------------
// Fused set-abstraction MLP: one block (4 waves) per (batch, center) group.
// Gather -> LDS, 3 WMMA layers ping-ponging in LDS, max-pool fused into the
// last layer via LDS int atomicMax on ReLU (>=0) bit patterns.
// ---------------------------------------------------------------------------
template<int NS, int CIN, int CF, int C0, int C1, int C2>
__global__ __launch_bounds__(128) void sa_fused_kernel(
    const float* __restrict__ xyz, int n, const half_t* __restrict__ feat,
    const float* __restrict__ new_xyz, const int* __restrict__ idx,
    const half_t* __restrict__ w0, const float* __restrict__ b0,
    const half_t* __restrict__ w1, const float* __restrict__ b1,
    const half_t* __restrict__ w2, const float* __restrict__ b2,
    half_t* __restrict__ out, int S) {
  constexpr int CA = (CIN > C1) ? CIN : C1;
  __shared__ alignas(16) half_t bufA[NS * CA];
  __shared__ alignas(16) half_t bufB[NS * C0];
  __shared__ int smax[C2];

  const int g = blockIdx.x;
  const int b = g / S;
  const int tid = threadIdx.x;
  const int wave = tid >> 5, lane = tid & 31;
  const int colL = lane & 15, hiL = lane >> 4;
  const int* gidx = idx + (size_t)g * NS;
  const float cx = new_xyz[(size_t)g * 3 + 0];
  const float cy = new_xyz[(size_t)g * 3 + 1];
  const float cz = new_xyz[(size_t)g * 3 + 2];

  for (int c = tid; c < C2; c += 128) smax[c] = 0;
  for (int e = tid; e < NS * CIN; e += 128) {
    int j = e / CIN, c = e % CIN;
    int p = gidx[j];
    float v = 0.f;
    if (c == 0)      v = xyz[((size_t)b * n + p) * 3 + 0] - cx;
    else if (c == 1) v = xyz[((size_t)b * n + p) * 3 + 1] - cy;
    else if (c == 2) v = xyz[((size_t)b * n + p) * 3 + 2] - cz;
    else {
      if constexpr (CF > 0) {
        if (c - 3 < CF) v = (float)feat[((size_t)b * n + p) * CF + (c - 3)];
      }
    }
    bufA[e] = (half_t)v;
  }
  __syncthreads();

  // layer 0: bufA (stride CIN) -> bufB (stride C0)
  {
    constexpr int RT = NS / 16, CT = C0 / 16;
    for (int t = wave; t < RT * CT; t += 4) {
      int rt = t / CT, ct = t % CT;
      v8f acc = {};
#pragma unroll
      for (int k = 0; k < CIN; k += 32)
        acc = wmma_step(load_a_frag(bufA + rt * 16 * CIN + k, CIN),
                        load_b_frag(w0 + (size_t)ct * 16 * CIN + k, CIN), acc);
      float bv = b0[ct * 16 + colL];
#pragma unroll
      for (int r = 0; r < 8; ++r)
        bufB[(rt * 16 + r + 8 * hiL) * C0 + ct * 16 + colL] =
            (half_t)fmaxf(acc[r] + bv, 0.f);
    }
  }
  __syncthreads();

  // layer 1: bufB (stride C0) -> bufA (stride C1)
  {
    constexpr int RT = NS / 16, CT = C1 / 16;
    for (int t = wave; t < RT * CT; t += 4) {
      int rt = t / CT, ct = t % CT;
      v8f acc = {};
#pragma unroll
      for (int k = 0; k < C0; k += 32)
        acc = wmma_step(load_a_frag(bufB + rt * 16 * C0 + k, C0),
                        load_b_frag(w1 + (size_t)ct * 16 * C0 + k, C0), acc);
      float bv = b1[ct * 16 + colL];
#pragma unroll
      for (int r = 0; r < 8; ++r)
        bufA[(rt * 16 + r + 8 * hiL) * C1 + ct * 16 + colL] =
            (half_t)fmaxf(acc[r] + bv, 0.f);
    }
  }
  __syncthreads();

  // layer 2 + fused max-pool over NS samples
  {
    constexpr int RT = NS / 16, CT = C2 / 16;
    for (int t = wave; t < RT * CT; t += 4) {
      int rt = t / CT, ct = t % CT;
      v8f acc = {};
#pragma unroll
      for (int k = 0; k < C1; k += 32)
        acc = wmma_step(load_a_frag(bufA + rt * 16 * C1 + k, C1),
                        load_b_frag(w2 + (size_t)ct * 16 * C1 + k, C1), acc);
      float bv = b2[ct * 16 + colL];
      float m = 0.f;
#pragma unroll
      for (int r = 0; r < 8; ++r) m = fmaxf(m, acc[r] + bv);
      m = fmaxf(m, 0.f);                      // ReLU before max; >= 0
      m = fmaxf(m, __shfl_xor(m, 16));        // combine the two 8-row halves
      if (lane < 16) atomicMax(&smax[ct * 16 + lane], __float_as_int(m));
    }
  }
  __syncthreads();
  for (int c = tid; c < C2; c += 128)
    out[(size_t)g * C2 + c] = (half_t)__int_as_float(smax[c]);
}

__global__ void concat_sa3_kernel(const float* __restrict__ xyz,
                                  const half_t* __restrict__ feat,
                                  half_t* __restrict__ out) {
  int t = blockIdx.x * blockDim.x + threadIdx.x;
  if (t >= 32 * 128 * 288) return;
  int row = t / 288, c = t % 288;
  float v = 0.f;
  if (c < 3)        v = xyz[(size_t)row * 3 + c];
  else if (c < 259) v = (float)feat[(size_t)row * 256 + (c - 3)];
  out[t] = (half_t)v;
}

// ---------------------------------------------------------------------------
// Generic WMMA GEMM: out[M,N] f16 = act(A[M,K] f16 x W[N,K]^T f16 + bias)
// Block = 4 waves, each wave owns one 16x16 tile of a 16-row x 64-col strip.
// ---------------------------------------------------------------------------
__global__ __launch_bounds__(128) void gemm_kernel(const half_t* __restrict__ A,
                                                   const half_t* __restrict__ W,
                                                   const float* __restrict__ bias,
                                                   half_t* __restrict__ out,
                                                   int M, int K, int N, int Nb, int relu) {
  const int wave = threadIdx.x >> 5, lane = threadIdx.x & 31;
  const int colL = lane & 15, hiL = lane >> 4;
  const int rt = blockIdx.x;
  const int ct = blockIdx.y * 4 + wave;
  if (ct * 16 >= N) return;
  const half_t* Ar = A + (size_t)rt * 16 * K;
  const half_t* Wr = W + (size_t)ct * 16 * K;
  v8f acc = {};
  for (int k = 0; k < K; k += 32)
    acc = wmma_step(load_a_frag(Ar + k, K), load_b_frag(Wr + k, K), acc);
  int col = ct * 16 + colL;
  float bv = (col < Nb) ? bias[col] : 0.f;
#pragma unroll
  for (int r = 0; r < 8; ++r) {
    float v = acc[r] + bv;
    if (relu) v = fmaxf(v, 0.f);
    out[(size_t)(rt * 16 + r + 8 * hiL) * N + col] = (half_t)v;
  }
}

__global__ void maxpool_kernel(const half_t* __restrict__ in, half_t* __restrict__ out,
                               int S, int C) {
  int t = blockIdx.x * blockDim.x + threadIdx.x;
  if (t >= 32 * C) return;
  int b = t / C, c = t % C;
  const half_t* p = in + (size_t)b * S * C + c;
  float m = 0.f;  // inputs are post-ReLU (>= 0)
  for (int j = 0; j < S; ++j) m = fmaxf(m, (float)p[(size_t)j * C]);
  out[t] = (half_t)m;
}

__global__ void softmax_kernel(const half_t* __restrict__ logits, float* __restrict__ out) {
  int b = threadIdx.x;
  if (b >= 32) return;
  float v[40], mx = -1e30f;
  for (int c = 0; c < 40; ++c) { v[c] = (float)logits[b * 48 + c]; mx = fmaxf(mx, v[c]); }
  float s = 0.f;
  for (int c = 0; c < 40; ++c) { v[c] = __expf(v[c] - mx); s += v[c]; }
  float inv = 1.f / s;
  for (int c = 0; c < 40; ++c) out[b * 40 + c] = v[c] * inv;
}

// ---------------------------------------------------------------------------
// Host orchestration
// ---------------------------------------------------------------------------
extern "C" void kernel_launch(void* const* d_in, const int* in_sizes, int n_in,
                              void* d_out, int out_size, void* d_ws, size_t ws_size,
                              hipStream_t stream) {
  (void)in_sizes; (void)n_in; (void)out_size; (void)ws_size;
  const float* x       = (const float*)d_in[0];
  const float* sa1_w0  = (const float*)d_in[1];  const float* sa1_b0 = (const float*)d_in[2];
  const float* sa1_w1  = (const float*)d_in[3];  const float* sa1_b1 = (const float*)d_in[4];
  const float* sa1_w2  = (const float*)d_in[5];  const float* sa1_b2 = (const float*)d_in[6];
  const float* sa2_w0  = (const float*)d_in[7];  const float* sa2_b0 = (const float*)d_in[8];
  const float* sa2_w1  = (const float*)d_in[9];  const float* sa2_b1 = (const float*)d_in[10];
  const float* sa2_w2  = (const float*)d_in[11]; const float* sa2_b2 = (const float*)d_in[12];
  const float* sa3_w0  = (const float*)d_in[13]; const float* sa3_b0 = (const float*)d_in[14];
  const float* sa3_w1  = (const float*)d_in[15]; const float* sa3_b1 = (const float*)d_in[16];
  const float* sa3_w2  = (const float*)d_in[17]; const float* sa3_b2 = (const float*)d_in[18];
  const float* fc1_w   = (const float*)d_in[19]; const float* fc1_b  = (const float*)d_in[20];
  const float* fc2_w   = (const float*)d_in[21]; const float* fc2_b  = (const float*)d_in[22];
  const float* fc3_w   = (const float*)d_in[23]; const float* fc3_b  = (const float*)d_in[24];

  char* ws = (char*)d_ws;
  size_t off = 0;
  auto alloc = [&](size_t bytes) -> void* {
    void* p = ws + off;
    off += (bytes + 255) & ~(size_t)255;
    return p;
  };

  // padded f16 weights
  half_t* sa1w0p = (half_t*)alloc((size_t)64 * 32 * 2);
  half_t* sa1w1p = (half_t*)alloc((size_t)64 * 64 * 2);
  half_t* sa1w2p = (half_t*)alloc((size_t)128 * 64 * 2);
  half_t* sa2w0p = (half_t*)alloc((size_t)128 * 160 * 2);
  half_t* sa2w1p = (half_t*)alloc((size_t)128 * 128 * 2);
  half_t* sa2w2p = (half_t*)alloc((size_t)256 * 128 * 2);
  half_t* sa3w0p = (half_t*)alloc((size_t)256 * 288 * 2);
  half_t* sa3w1p = (half_t*)alloc((size_t)512 * 256 * 2);
  half_t* sa3w2p = (half_t*)alloc((size_t)1024 * 512 * 2);
  half_t* fc1p   = (half_t*)alloc((size_t)512 * 1024 * 2);
  half_t* fc2p   = (half_t*)alloc((size_t)256 * 512 * 2);
  half_t* fc3p   = (half_t*)alloc((size_t)48 * 256 * 2);
  // stage buffers
  int*    fps1_idx = (int*)alloc((size_t)32 * 512 * 4);
  float*  new_xyz1 = (float*)alloc((size_t)32 * 512 * 3 * 4);
  int*    bq1_idx  = (int*)alloc((size_t)32 * 512 * 32 * 4);
  half_t* feat1    = (half_t*)alloc((size_t)32 * 512 * 128 * 2);
  int*    fps2_idx = (int*)alloc((size_t)32 * 128 * 4);
  float*  new_xyz2 = (float*)alloc((size_t)32 * 128 * 3 * 4);
  int*    bq2_idx  = (int*)alloc((size_t)32 * 128 * 64 * 4);
  half_t* feat2    = (half_t*)alloc((size_t)32 * 128 * 256 * 2);
  half_t* sa3_in   = (half_t*)alloc((size_t)4096 * 288 * 2);
  half_t* sa3_t0   = (half_t*)alloc((size_t)4096 * 256 * 2);
  half_t* sa3_t1   = (half_t*)alloc((size_t)4096 * 512 * 2);
  half_t* sa3_t2   = (half_t*)alloc((size_t)4096 * 1024 * 2);
  half_t* gfeat    = (half_t*)alloc((size_t)32 * 1024 * 2);
  half_t* fct1     = (half_t*)alloc((size_t)32 * 512 * 2);
  half_t* fct2     = (half_t*)alloc((size_t)32 * 256 * 2);
  half_t* logits   = (half_t*)alloc((size_t)32 * 48 * 2);

  auto convw = [&](const float* w, half_t* dst, int O, int I, int Op, int Ip) {
    int tot = Op * Ip;
    convert_w_kernel<<<(tot + 255) / 256, 256, 0, stream>>>(w, dst, O, I, Op, Ip);
  };
  convw(sa1_w0, sa1w0p, 64, 3, 64, 32);
  convw(sa1_w1, sa1w1p, 64, 64, 64, 64);
  convw(sa1_w2, sa1w2p, 128, 64, 128, 64);
  convw(sa2_w0, sa2w0p, 128, 131, 128, 160);
  convw(sa2_w1, sa2w1p, 128, 128, 128, 128);
  convw(sa2_w2, sa2w2p, 256, 128, 256, 128);
  convw(sa3_w0, sa3w0p, 256, 259, 256, 288);
  convw(sa3_w1, sa3w1p, 512, 256, 512, 256);
  convw(sa3_w2, sa3w2p, 1024, 512, 1024, 512);
  convw(fc1_w, fc1p, 512, 1024, 512, 1024);
  convw(fc2_w, fc2p, 256, 512, 256, 512);
  convw(fc3_w, fc3p, 40, 256, 48, 256);

  // ---- SA1: FPS(4096->512), ball(r=0.2, ns=32), MLP [64,64,128] + maxpool
  fps_kernel<16><<<32, 256, 0, stream>>>(x, 512, fps1_idx);
  gather_xyz_kernel<<<(32 * 512 + 255) / 256, 256, 0, stream>>>(x, fps1_idx, new_xyz1, 4096, 512);
  ball_query_kernel<4096><<<32, 256, 0, stream>>>(new_xyz1, x, 512, 32, 0.04f, bq1_idx);
  sa_fused_kernel<32, 32, 0, 64, 64, 128><<<32 * 512, 128, 0, stream>>>(
      x, 4096, nullptr, new_xyz1, bq1_idx,
      sa1w0p, sa1_b0, sa1w1p, sa1_b1, sa1w2p, sa1_b2, feat1, 512);

  // ---- SA2: FPS(512->128), ball(r=0.4, ns=64), MLP [128,128,256] + maxpool
  fps_kernel<2><<<32, 256, 0, stream>>>(new_xyz1, 128, fps2_idx);
  gather_xyz_kernel<<<(32 * 128 + 255) / 256, 256, 0, stream>>>(new_xyz1, fps2_idx, new_xyz2, 512, 128);
  ball_query_kernel<512><<<32, 256, 0, stream>>>(new_xyz2, new_xyz1, 128, 64, 0.16f, bq2_idx);
  sa_fused_kernel<64, 160, 128, 128, 128, 256><<<32 * 128, 128, 0, stream>>>(
      new_xyz1, 512, feat1, new_xyz2, bq2_idx,
      sa2w0p, sa2_b0, sa2w1p, sa2_b1, sa2w2p, sa2_b2, feat2, 128);

  // ---- SA3 (group-all): concat -> 3 GEMMs (M=4096) -> maxpool over 128
  concat_sa3_kernel<<<(32 * 128 * 288 + 255) / 256, 256, 0, stream>>>(new_xyz2, feat2, sa3_in);
  gemm_kernel<<<dim3(256, 4),  128, 0, stream>>>(sa3_in, sa3w0p, sa3_b0, sa3_t0, 4096, 288, 256, 256, 1);
  gemm_kernel<<<dim3(256, 8),  128, 0, stream>>>(sa3_t0, sa3w1p, sa3_b1, sa3_t1, 4096, 256, 512, 512, 1);
  gemm_kernel<<<dim3(256, 16), 128, 0, stream>>>(sa3_t1, sa3w2p, sa3_b2, sa3_t2, 4096, 512, 1024, 1024, 1);
  maxpool_kernel<<<(32 * 1024 + 255) / 256, 256, 0, stream>>>(sa3_t2, gfeat, 128, 1024);

  // ---- FC head + softmax
  gemm_kernel<<<dim3(2, 8), 128, 0, stream>>>(gfeat, fc1p, fc1_b, fct1, 32, 1024, 512, 512, 1);
  gemm_kernel<<<dim3(2, 4), 128, 0, stream>>>(fct1, fc2p, fc2_b, fct2, 32, 512, 256, 256, 1);
  gemm_kernel<<<dim3(2, 1), 128, 0, stream>>>(fct2, fc3p, fc3_b, logits, 32, 256, 48, 40, 0);
  softmax_kernel<<<1, 32, 0, stream>>>(logits, (float*)d_out);
}
